// TransLayer_81458349736311
// MI455X (gfx1250) — matrix operand order repaired and checked by reading, not compiled, not run on hardware
//
#include <hip/hip_runtime.h>

typedef __bf16 bf16;
typedef __attribute__((ext_vector_type(16))) __bf16 v16bf;
typedef __attribute__((ext_vector_type(8)))  float  v8f;

#define DEV __device__ __forceinline__

DEV int lane_id() { return threadIdx.x & 31; }

DEV v8f wmma_bf16(v16bf a, v16bf b, v8f c) {
  // v_wmma_f32_16x16x32_bf16: D(f32 16x16) = A(16x32 bf16) * B(32x16 bf16) + C
  return __builtin_amdgcn_wmma_f32_16x16x32_bf16(false, a, false, b, (short)0, c, false, false);
}

// CDNA5 async global->LDS copy (ASYNCcnt-tracked, no VGPR round trip).
// GV addressing: 64-bit vaddr, saddr = off. LDS byte address = low 32 bits of
// the generic shared pointer.
DEV void async_b128(void* lds, const void* g) {
  asm volatile("global_load_async_to_lds_b128 %0, %1, off"
               :: "v"((unsigned)(unsigned long long)lds), "v"(g) : "memory");
}
DEV void wait_async0() { asm volatile("s_wait_asynccnt 0" ::: "memory"); }

// A fragment: 16x32 (MxK) bf16 from row-major [M][K] memory (ld elems).
// Lane L<16: row M=L, k = {0..7, 16..23}; lane>=16: row M=L-16, k = {8..15, 24..31}.
DEV v16bf load_frag_a(const bf16* p, int ld) {
  int lane = lane_id();
  const bf16* r = p + (lane & 15) * ld + ((lane >> 4) & 1) * 8;
  v16bf f;
#pragma unroll
  for (int e = 0; e < 8; ++e) f[e] = r[e];
#pragma unroll
  for (int e = 0; e < 8; ++e) f[8 + e] = r[16 + e];
  return f;
}

// B fragment: 32x16 (KxN) bf16, memory stored as [N][K] row-major (ld elems).
// Lane L<16: col N=L, k = {0..7, 16..23}; lane>=16: col N=L-16, k = {8..15, 24..31}.
DEV v16bf load_frag_b(const bf16* p, int ld) {
  int lane = lane_id();
  const bf16* r = p + (lane & 15) * ld + ((lane >> 4) & 1) * 8;
  v16bf f;
#pragma unroll
  for (int e = 0; e < 8; ++e) f[e] = r[e];
#pragma unroll
  for (int e = 0; e < 8; ++e) f[8 + e] = r[16 + e];
  return f;
}

// C/D layout: vgpr e -> row = e + 8*(lane>=16), col = lane&15
DEV int frag_row(int e) { return e + ((lane_id() >> 4) & 1) * 8; }
DEV int frag_col() { return lane_id() & 15; }

DEV float half_max(float v) {
#pragma unroll
  for (int m = 1; m < 16; m <<= 1) v = fmaxf(v, __shfl_xor(v, m, 32));
  return v;
}
DEV float half_sum(float v) {
#pragma unroll
  for (int m = 1; m < 16; m <<= 1) v += __shfl_xor(v, m, 32);
  return v;
}

// ---------------------------------------------------------------- LayerNorm
__global__ __launch_bounds__(256) void ln_kernel(const float* __restrict__ x,
    const float* __restrict__ g, const float* __restrict__ b,
    bf16* __restrict__ out, int rows) {
  int wave = threadIdx.x >> 5, lane = threadIdx.x & 31;
  int r = blockIdx.x * 8 + wave;
  if (r >= rows) return;
  const float* xr = x + (size_t)r * 512;
  float v[16]; float s = 0.f;
#pragma unroll
  for (int i = 0; i < 16; ++i) { v[i] = xr[lane + 32 * i]; s += v[i]; }
#pragma unroll
  for (int m = 1; m < 32; m <<= 1) s += __shfl_xor(s, m, 32);
  float mu = s * (1.f / 512.f);
  float q = 0.f;
#pragma unroll
  for (int i = 0; i < 16; ++i) { float d = v[i] - mu; q += d * d; }
#pragma unroll
  for (int m = 1; m < 32; m <<= 1) q += __shfl_xor(q, m, 32);
  float rstd = rsqrtf(q * (1.f / 512.f) + 1e-5f);
  bf16* o = out + (size_t)r * 512;
#pragma unroll
  for (int i = 0; i < 16; ++i) {
    int c = lane + 32 * i;
    o[c] = (bf16)((v[i] - mu) * rstd * g[c] + b[c]);
  }
}

__global__ void cast_f32_to_bf16(const float* __restrict__ s, bf16* __restrict__ d, int n) {
  int i = blockIdx.x * 256 + threadIdx.x;
  if (i < n) d[i] = (bf16)s[i];
}

// ------------------------------------------- big GEMM (QKV and output proj)
// C[M x N] = A[M][512] @ W[N][512]^T ; BM=BN=128, Kc=64, 8 waves (2x4), wave=64x32
// Double-buffered LDS staged via async global->LDS (ASYNCcnt), overlapping
// the next K-step's 36KB stage with the current step's WMMA work.
__global__ __launch_bounds__(256) void gemm_xw_kernel(
    const bf16* __restrict__ A, const bf16* __restrict__ W,
    bf16* __restrict__ qo, bf16* __restrict__ ko, bf16* __restrict__ vo,
    const float* __restrict__ xres, const float* __restrict__ bias,
    float* __restrict__ yo, int mode) {
  extern __shared__ char smemc[];                 // 2 x (128x72 A + 128x72 B)
  int tid = threadIdx.x, wave = tid >> 5;
  int m0 = blockIdx.y * 128, n0 = blockIdx.x * 128;
  int wm = wave >> 2, wn = wave & 3;
  v8f acc[4][2] = {};

  auto stage = [&](int k0, int buf) {
    bf16* dA = (bf16*)(smemc + buf * 36864);
    bf16* dB = dA + 128 * 72;
#pragma unroll
    for (int i = 0; i < 4; ++i) {
      int c = tid + i * 256;
      int r = c >> 3, c8 = (c & 7) * 8;
      async_b128(dA + r * 72 + c8, A + ((size_t)m0 + r) * 512 + k0 + c8);
    }
#pragma unroll
    for (int i = 0; i < 4; ++i) {
      int c = tid + i * 256;
      int r = c >> 3, c8 = (c & 7) * 8;
      async_b128(dB + r * 72 + c8, W + ((size_t)n0 + r) * 512 + k0 + c8);
    }
  };

  stage(0, 0);
  wait_async0();
  __syncthreads();
  int nbuf = 0;
  for (int ks = 0; ks < 8; ++ks) {
    if (ks + 1 < 8) stage((ks + 1) * 64, nbuf ^ 1);
    bf16* sA = (bf16*)(smemc + nbuf * 36864);
    bf16* sB = sA + 128 * 72;
#pragma unroll
    for (int kk = 0; kk < 2; ++kk) {
      v16bf afr[4], bfr[2];
#pragma unroll
      for (int i = 0; i < 4; ++i) afr[i] = load_frag_a(&sA[(wm * 64 + i * 16) * 72 + kk * 32], 72);
#pragma unroll
      for (int j = 0; j < 2; ++j) bfr[j] = load_frag_b(&sB[(wn * 32 + j * 16) * 72 + kk * 32], 72);
#pragma unroll
      for (int i = 0; i < 4; ++i)
#pragma unroll
        for (int j = 0; j < 2; ++j) acc[i][j] = wmma_bf16(afr[i], bfr[j], acc[i][j]);
    }
    wait_async0();
    __syncthreads();
    nbuf ^= 1;
  }
#pragma unroll
  for (int i = 0; i < 4; ++i)
#pragma unroll
    for (int j = 0; j < 2; ++j) {
      int n = n0 + wn * 32 + j * 16 + frag_col();
#pragma unroll
      for (int e = 0; e < 8; ++e) {
        int m = m0 + wm * 64 + i * 16 + frag_row(e);
        float val = acc[i][j][e];
        if (mode == 0) {
          int which = n >> 9, c = n & 511, h = c >> 6, d = c & 63;
          int bb = m >> 13, row = m & 8191;
          size_t idx = (((size_t)bb * 8 + h) * 8192 + row) * 64 + d;
          if (which == 0) qo[idx] = (bf16)(val * 0.125f);   // * dh^-0.5
          else if (which == 1) ko[idx] = (bf16)val;
          else vo[idx] = (bf16)val;
        } else {
          yo[(size_t)m * 512 + n] = val + xres[(size_t)m * 512 + n] + bias[n];
        }
      }
    }
}

// ------------------------------------------------------------- landmarks
__global__ void landmark_kernel(const bf16* __restrict__ q, const bf16* __restrict__ k,
                                bf16* __restrict__ ql, bf16* __restrict__ kl) {
  int bh = blockIdx.x;
  int lm = blockIdx.y * 4 + (threadIdx.x >> 6), d = threadIdx.x & 63;
  const bf16* qb = q + ((size_t)bh * 8192 + lm * 32) * 64 + d;
  const bf16* kb = k + ((size_t)bh * 8192 + lm * 32) * 64 + d;
  float sq = 0.f, sk = 0.f;
#pragma unroll
  for (int j = 0; j < 32; ++j) { sq += (float)qb[j * 64]; sk += (float)kb[j * 64]; }
  size_t o = ((size_t)bh * 256 + lm) * 64 + d;
  ql[o] = (bf16)(sq * (1.f / 32.f));
  kl[o] = (bf16)(sk * (1.f / 32.f));
}

// --------------------------------------------- sim2 = ql@kl^T (raw logits)
__global__ __launch_bounds__(256) void sim2_gemm_kernel(const bf16* __restrict__ ql,
    const bf16* __restrict__ kl, float* __restrict__ attn2) {
  extern __shared__ char smemc[];
  bf16* sq = (bf16*)smemc;                 // 128 x 72
  bf16* sk = (bf16*)(smemc + 128 * 72 * 2);// 256 x 72
  int bh = blockIdx.y, tid = threadIdx.x, wave = tid >> 5;
  const bf16* qg = ql + (size_t)bh * 256 * 64 + (size_t)blockIdx.x * 128 * 64;
  const bf16* kg = kl + (size_t)bh * 256 * 64;
  for (int i = 0; i < 32; ++i) { int l = tid + i * 256, r = l >> 6, c = l & 63; sq[r * 72 + c] = qg[r * 64 + c]; }
  for (int i = 0; i < 64; ++i) { int l = tid + i * 256, r = l >> 6, c = l & 63; sk[r * 72 + c] = kg[r * 64 + c]; }
  __syncthreads();
  v16bf af[2];
#pragma unroll
  for (int kk = 0; kk < 2; ++kk) af[kk] = load_frag_a(&sq[(wave * 16) * 72 + kk * 32], 72);
  float* orow = attn2 + (size_t)bh * 65536 + (size_t)blockIdx.x * 128 * 256;
  for (int ni = 0; ni < 16; ++ni) {
    v8f s = {};
#pragma unroll
    for (int kk = 0; kk < 2; ++kk) {
      v16bf bfr = load_frag_b(&sk[(ni * 16) * 72 + kk * 32], 72);
      s = wmma_bf16(af[kk], bfr, s);
    }
    int col = ni * 16 + frag_col();
#pragma unroll
    for (int e = 0; e < 8; ++e) orow[(wave * 16 + frag_row(e)) * 256 + col] = s[e];
  }
}

// softmax rows of attn2 in place + bf16 copy + per-bh max column-sum
__global__ void softmax256_kernel(float* __restrict__ attn2, bf16* __restrict__ abf,
                                  float* __restrict__ bhmax) {
  __shared__ float red[256];
  int bh = blockIdx.x, r = threadIdx.x;
  float* a = attn2 + (size_t)bh * 65536;
  float* p = a + r * 256;
  float mx = -1e30f;
  for (int j = 0; j < 256; ++j) mx = fmaxf(mx, p[j]);
  float s = 0.f;
  for (int j = 0; j < 256; ++j) s += __expf(p[j] - mx);
  float inv = 1.f / s;
  bf16* ab = abf + (size_t)bh * 65536 + (size_t)r * 256;
  for (int j = 0; j < 256; ++j) { float v = __expf(p[j] - mx) * inv; p[j] = v; ab[j] = (bf16)v; }
  __syncthreads();
  float cs = 0.f;
  for (int i = 0; i < 256; ++i) cs += a[i * 256 + r];
  red[r] = cs; __syncthreads();
  for (int st = 128; st > 0; st >>= 1) { if (r < st) red[r] = fmaxf(red[r], red[r + st]); __syncthreads(); }
  if (r == 0) bhmax[bh] = red[0];
}

// col = max row-sum of softmax == 1 exactly; row = global max col-sum
__global__ void rowscale_kernel(const float* __restrict__ bhmax, float* __restrict__ invrow) {
  float v = bhmax[threadIdx.x];
#pragma unroll
  for (int m = 1; m < 32; m <<= 1) v = fmaxf(v, __shfl_xor(v, m, 32));
  if (threadIdx.x == 0) invrow[0] = 1.0f / v;
}

// z0 = attn2^T * invrow  (bf16)
__global__ void zinit_kernel(const float* __restrict__ attn2, const float* __restrict__ invrow,
                             bf16* __restrict__ z) {
  int i = blockIdx.x * 256 + threadIdx.x;           // 32*65536
  int bh = i >> 16, rc = i & 65535, r = rc >> 8, c = rc & 255;
  z[i] = (bf16)(attn2[(size_t)bh * 65536 + c * 256 + r] * invrow[0]);
}

// ------------------- batched 256-GEMM: C = alpha*A@B + delta*A  (32 batches)
// A:[256][256], B:[256][N] row-major, C:[256][N]. BM=128, BN=64, Kc=64.
__global__ __launch_bounds__(256) void bgemm_kernel(const bf16* __restrict__ A,
    const bf16* __restrict__ B, bf16* __restrict__ C, int N, float alpha, float delta) {
  extern __shared__ char smemc[];
  bf16* sA = (bf16*)smemc;                  // 128 x 72
  bf16* sBt = (bf16*)(smemc + 128 * 72 * 2);// 64 x 72 (transposed)
  const int K = 256;
  int bh = blockIdx.z;
  A += (size_t)bh * 256 * K; B += (size_t)bh * (size_t)K * N; C += (size_t)bh * 256 * N;
  int m0 = blockIdx.y * 128, n0 = blockIdx.x * 64;
  int tid = threadIdx.x, wave = tid >> 5;
  int wm = wave >> 1, wn = wave & 1;
  v8f acc[2][2] = {};
  for (int k0 = 0; k0 < K; k0 += 64) {
    for (int i = 0; i < 32; ++i) { int l = tid + i * 256, r = l >> 6, c = l & 63; sA[r * 72 + c] = A[(m0 + r) * K + k0 + c]; }
    for (int i = 0; i < 16; ++i) { int l = tid + i * 256, k = l >> 6, n = l & 63; sBt[n * 72 + k] = B[(k0 + k) * N + n0 + n]; }
    __syncthreads();
#pragma unroll
    for (int kk = 0; kk < 2; ++kk) {
      v16bf a0 = load_frag_a(&sA[(wm * 32) * 72 + kk * 32], 72);
      v16bf a1 = load_frag_a(&sA[(wm * 32 + 16) * 72 + kk * 32], 72);
      v16bf b0 = load_frag_b(&sBt[(wn * 32) * 72 + kk * 32], 72);
      v16bf b1 = load_frag_b(&sBt[(wn * 32 + 16) * 72 + kk * 32], 72);
      acc[0][0] = wmma_bf16(a0, b0, acc[0][0]);
      acc[0][1] = wmma_bf16(a0, b1, acc[0][1]);
      acc[1][0] = wmma_bf16(a1, b0, acc[1][0]);
      acc[1][1] = wmma_bf16(a1, b1, acc[1][1]);
    }
    __syncthreads();
  }
#pragma unroll
  for (int mi = 0; mi < 2; ++mi)
#pragma unroll
    for (int ni = 0; ni < 2; ++ni) {
      int col = n0 + wn * 32 + ni * 16 + frag_col();
#pragma unroll
      for (int e = 0; e < 8; ++e) {
        int row = m0 + wm * 32 + mi * 16 + frag_row(e);
        float v = alpha * acc[mi][ni][e];
        if (delta != 0.f) v += delta * (float)A[row * K + col];
        C[row * N + col] = (bf16)v;
      }
    }
}

// ------------------ flash pass A: softmax stats of sim3 = ql@k^T (split-K)
__global__ __launch_bounds__(256) void flash3_stats_kernel(const bf16* __restrict__ ql,
    const bf16* __restrict__ k, float* __restrict__ pm, float* __restrict__ pl) {
  extern __shared__ char smemc[];
  bf16* sq = (bf16*)smemc;                  // 256 x 72
  bf16* sk = (bf16*)(smemc + 256 * 72 * 2); // 128 x 72
  int chunk = blockIdx.x, bh = blockIdx.y, tid = threadIdx.x, wave = tid >> 5;
  const bf16* qg = ql + (size_t)bh * 256 * 64;
  const bf16* kg = k + ((size_t)bh * 8192 + chunk * 1024) * 64;
  for (int i = 0; i < 64; ++i) { int l = tid + i * 256, r = l >> 6, c = l & 63; sq[r * 72 + c] = qg[r * 64 + c]; }
  __syncthreads();
  v16bf af[2][2];
#pragma unroll
  for (int mi = 0; mi < 2; ++mi)
#pragma unroll
    for (int kk = 0; kk < 2; ++kk) af[mi][kk] = load_frag_a(&sq[(wave * 32 + mi * 16) * 72 + kk * 32], 72);
  float m_run[2][8], l_run[2][8];
#pragma unroll
  for (int mi = 0; mi < 2; ++mi)
#pragma unroll
    for (int e = 0; e < 8; ++e) { m_run[mi][e] = -1e30f; l_run[mi][e] = 0.f; }
  for (int st = 0; st < 8; ++st) {
    __syncthreads();
    for (int i = 0; i < 32; ++i) { int l = tid + i * 256, r = l >> 6, c = l & 63; sk[r * 72 + c] = kg[(st * 128 + r) * 64 + c]; }
    __syncthreads();
    v8f s[2][8] = {};
    for (int ni = 0; ni < 8; ++ni)
#pragma unroll
      for (int kk = 0; kk < 2; ++kk) {
        v16bf bfr = load_frag_b(&sk[(ni * 16) * 72 + kk * 32], 72);
        s[0][ni] = wmma_bf16(af[0][kk], bfr, s[0][ni]);
        s[1][ni] = wmma_bf16(af[1][kk], bfr, s[1][ni]);
      }
#pragma unroll
    for (int mi = 0; mi < 2; ++mi)
#pragma unroll
      for (int e = 0; e < 8; ++e) {
        float mx = -1e30f;
#pragma unroll
        for (int ni = 0; ni < 8; ++ni) mx = fmaxf(mx, s[mi][ni][e]);
        mx = half_max(mx);
        float mnew = fmaxf(m_run[mi][e], mx);
        float ssum = 0.f;
#pragma unroll
        for (int ni = 0; ni < 8; ++ni) ssum += __expf(s[mi][ni][e] - mnew);
        ssum = half_sum(ssum);
        l_run[mi][e] = l_run[mi][e] * __expf(m_run[mi][e] - mnew) + ssum;
        m_run[mi][e] = mnew;
      }
  }
  if ((tid & 15) == 0) {
    size_t base = ((size_t)chunk * 32 + bh) * 256;
#pragma unroll
    for (int mi = 0; mi < 2; ++mi)
#pragma unroll
      for (int e = 0; e < 8; ++e) {
        int row = wave * 32 + mi * 16 + e + ((tid >> 4) & 1) * 8;
        pm[base + row] = m_run[mi][e];
        pl[base + row] = l_run[mi][e];
      }
  }
}

__global__ void stat_combine_kernel(const float* __restrict__ pm, const float* __restrict__ pl,
                                    float* __restrict__ mf, float* __restrict__ lf) {
  int bh = blockIdx.x, r = threadIdx.x;
  float m = -1e30f;
  for (int c = 0; c < 8; ++c) m = fmaxf(m, pm[((size_t)c * 32 + bh) * 256 + r]);
  float l = 0.f;
  for (int c = 0; c < 8; ++c) l += pl[((size_t)c * 32 + bh) * 256 + r] * __expf(pm[((size_t)c * 32 + bh) * 256 + r] - m);
  mf[bh * 256 + r] = m;
  lf[bh * 256 + r] = fmaxf(l, 1e-30f);
}

// ------ flash pass B: partial O = softmax(ql@k^T) @ v over key chunk (split-K)
__global__ __launch_bounds__(256) void flash3_av_kernel(const bf16* __restrict__ ql,
    const bf16* __restrict__ k, const bf16* __restrict__ v,
    const float* __restrict__ mf, const float* __restrict__ lf, float* __restrict__ pO) {
  extern __shared__ char smemc[];
  bf16* sq = (bf16*)smemc;                                  // 256 x 72
  bf16* sk = (bf16*)(smemc + 256 * 72 * 2);                 // 128 x 72
  bf16* sP = (bf16*)(smemc + (256 + 128) * 72 * 2);         // 256 x 136
  bf16* sv = (bf16*)(smemc + (256 + 128) * 72 * 2 + 256 * 136 * 2); // 64 x 136
  int chunk = blockIdx.x, bh = blockIdx.y, tid = threadIdx.x, wave = tid >> 5, lane = tid & 31;
  const bf16* qg = ql + (size_t)bh * 256 * 64;
  const bf16* kg = k + ((size_t)bh * 8192 + chunk * 1024) * 64;
  const bf16* vg = v + ((size_t)bh * 8192 + chunk * 1024) * 64;
  for (int i = 0; i < 64; ++i) { int l = tid + i * 256, r = l >> 6, c = l & 63; sq[r * 72 + c] = qg[r * 64 + c]; }
  __syncthreads();
  v16bf af[2][2];
#pragma unroll
  for (int mi = 0; mi < 2; ++mi)
#pragma unroll
    for (int kk = 0; kk < 2; ++kk) af[mi][kk] = load_frag_a(&sq[(wave * 32 + mi * 16) * 72 + kk * 32], 72);
  float mrow[2][8], linv[2][8];
#pragma unroll
  for (int mi = 0; mi < 2; ++mi)
#pragma unroll
    for (int e = 0; e < 8; ++e) {
      int row = wave * 32 + mi * 16 + e + ((lane >> 4) & 1) * 8;
      mrow[mi][e] = mf[bh * 256 + row];
      linv[mi][e] = 1.f / lf[bh * 256 + row];
    }
  v8f o[2][4] = {};
  for (int st = 0; st < 8; ++st) {
    __syncthreads();
    for (int i = 0; i < 32; ++i) { int l = tid + i * 256, r = l >> 6, c = l & 63; sk[r * 72 + c] = kg[(st * 128 + r) * 64 + c]; }
    for (int i = 0; i < 32; ++i) { int l = tid + i * 256, j = l >> 6, d = l & 63; sv[d * 136 + j] = vg[(st * 128 + j) * 64 + d]; }
    __syncthreads();
    for (int ni = 0; ni < 8; ++ni) {
      v8f s0 = {}, s1 = {};
#pragma unroll
      for (int kk = 0; kk < 2; ++kk) {
        v16bf bfr = load_frag_b(&sk[(ni * 16) * 72 + kk * 32], 72);
        s0 = wmma_bf16(af[0][kk], bfr, s0);
        s1 = wmma_bf16(af[1][kk], bfr, s1);
      }
      int key = ni * 16 + (lane & 15);
#pragma unroll
      for (int e = 0; e < 8; ++e) {
        int r0 = wave * 32 + e + ((lane >> 4) & 1) * 8;
        sP[r0 * 136 + key] = (bf16)(__expf(s0[e] - mrow[0][e]) * linv[0][e]);
        int r1 = r0 + 16;
        sP[r1 * 136 + key] = (bf16)(__expf(s1[e] - mrow[1][e]) * linv[1][e]);
      }
    }
    __syncthreads();
#pragma unroll
    for (int kk = 0; kk < 4; ++kk) {
      v16bf pa0 = load_frag_a(&sP[(wave * 32) * 136 + kk * 32], 136);
      v16bf pa1 = load_frag_a(&sP[(wave * 32 + 16) * 136 + kk * 32], 136);
#pragma unroll
      for (int nd = 0; nd < 4; ++nd) {
        v16bf vb = load_frag_b(&sv[(nd * 16) * 136 + kk * 32], 136);
        o[0][nd] = wmma_bf16(pa0, vb, o[0][nd]);
        o[1][nd] = wmma_bf16(pa1, vb, o[1][nd]);
      }
    }
  }
  size_t base = ((size_t)chunk * 32 + bh) * 256;
#pragma unroll
  for (int mi = 0; mi < 2; ++mi)
#pragma unroll
    for (int nd = 0; nd < 4; ++nd) {
      int d = nd * 16 + (lane & 15);
#pragma unroll
      for (int e = 0; e < 8; ++e) {
        int row = wave * 32 + mi * 16 + e + ((lane >> 4) & 1) * 8;
        pO[(base + row) * 64 + d] = o[mi][nd][e];
      }
    }
}

__global__ void sum_partials_kernel(const float* __restrict__ pO, bf16* __restrict__ av) {
  int i = blockIdx.x * 256 + threadIdx.x;  // 32*256*64
  float s = 0.f;
#pragma unroll
  for (int c = 0; c < 8; ++c) s += pO[(size_t)c * 524288 + i];
  av[i] = (bf16)s;
}

__global__ void transpose_p_kernel(const bf16* __restrict__ P, bf16* __restrict__ Pt) {
  int i = blockIdx.x * 256 + threadIdx.x;  // [bh][d][j] <- [bh][j][d]
  int bh = i >> 14, rem = i & 16383, d = rem >> 8, j = rem & 255;
  Pt[i] = P[((size_t)bh * 256 + j) * 64 + d];
}

// --------- flash1: out = softmax(q@kl^T) @ P  + depthwise-conv33(v), -> ctx
__global__ __launch_bounds__(256) void flash1_kernel(const bf16* __restrict__ q,
    const bf16* __restrict__ kl, const bf16* __restrict__ Pt, const bf16* __restrict__ v,
    const float* __restrict__ wres, bf16* __restrict__ ctx) {
  extern __shared__ char smemc[];
  bf16* sP = (bf16*)smemc;                  // 128 x 264
  bf16* sv = (bf16*)(smemc + 128 * 264 * 2);// 160 x 64
  int rb0 = blockIdx.x * 128, bh = blockIdx.y, tid = threadIdx.x, wave = tid >> 5, lane = tid & 31;
  int b = bh >> 3, h = bh & 7;
  const bf16* vg = v + (size_t)bh * 8192 * 64;
  for (int i = 0; i < 40; ++i) {
    int l = tid + i * 256, r = l >> 6, c = l & 63;
    int g = rb0 - 16 + r;
    sv[r * 64 + c] = (g >= 0 && g < 8192) ? vg[(size_t)g * 64 + c] : (bf16)0.f;
  }
  const bf16* qg = q + ((size_t)bh * 8192 + rb0 + wave * 16) * 64;
  const bf16* klg = kl + (size_t)bh * 256 * 64;
  v16bf af[2];
#pragma unroll
  for (int kk = 0; kk < 2; ++kk) af[kk] = load_frag_a(qg + kk * 32, 64);
  v8f s[16] = {};
  for (int ni = 0; ni < 16; ++ni)
#pragma unroll
    for (int kk = 0; kk < 2; ++kk) {
      v16bf bfr = load_frag_b(klg + (ni * 16) * 64 + kk * 32, 64);
      s[ni] = wmma_bf16(af[kk], bfr, s[ni]);
    }
#pragma unroll
  for (int e = 0; e < 8; ++e) {
    float mx = -1e30f;
#pragma unroll
    for (int ni = 0; ni < 16; ++ni) mx = fmaxf(mx, s[ni][e]);
    mx = half_max(mx);
    float ssum = 0.f;
#pragma unroll
    for (int ni = 0; ni < 16; ++ni) ssum += __expf(s[ni][e] - mx);
    ssum = half_sum(ssum);
    float inv = 1.f / ssum;
    int row = wave * 16 + e + ((lane >> 4) & 1) * 8;
#pragma unroll
    for (int ni = 0; ni < 16; ++ni)
      sP[row * 264 + ni * 16 + (lane & 15)] = (bf16)(__expf(s[ni][e] - mx) * inv);
  }
  __syncthreads();
  const bf16* ptg = Pt + (size_t)bh * 64 * 256;
  v8f o[4] = {};
#pragma unroll
  for (int kk = 0; kk < 8; ++kk) {
    v16bf pa = load_frag_a(&sP[(wave * 16) * 264 + kk * 32], 264);
#pragma unroll
    for (int nd = 0; nd < 4; ++nd) {
      v16bf pb = load_frag_b(ptg + (nd * 16) * 256 + kk * 32, 256);
      o[nd] = wmma_bf16(pa, pb, o[nd]);
    }
  }
  float wl[33];
#pragma unroll
  for (int t = 0; t < 33; ++t) wl[t] = wres[h * 33 + t];
#pragma unroll
  for (int nd = 0; nd < 4; ++nd) {
    int d = nd * 16 + (lane & 15);
#pragma unroll
    for (int e = 0; e < 8; ++e) {
      int rloc = wave * 16 + e + ((lane >> 4) & 1) * 8;
      float r = 0.f;
      for (int t = 0; t < 33; ++t) r += wl[t] * (float)sv[(rloc + t) * 64 + d];
      ctx[(((size_t)b * 8192) + rb0 + rloc) * 512 + h * 64 + d] = (bf16)(o[nd][e] + r);
    }
  }
}

// ===========================================================================
extern "C" void kernel_launch(void* const* d_in, const int* in_sizes, int n_in,
                              void* d_out, int out_size, void* d_ws, size_t ws_size,
                              hipStream_t stream) {
  (void)in_sizes; (void)n_in; (void)out_size; (void)ws_size;
  const float* x     = (const float*)d_in[0];
  const float* gamma = (const float*)d_in[1];
  const float* beta  = (const float*)d_in[2];
  const float* w_qkv = (const float*)d_in[3];
  const float* w_out = (const float*)d_in[4];
  const float* b_out = (const float*)d_in[5];
  const float* w_res = (const float*)d_in[6];
  float* out = (float*)d_out;

  char* ws = (char*)d_ws; size_t off = 0;
  auto alloc = [&](size_t bytes) -> char* {
    char* p = ws + off; off = (off + bytes + 255) & ~(size_t)255; return p;
  };
  const size_t MN = 32768, BHND = (size_t)32 * 8192 * 64;
  bf16* xn    = (bf16*)alloc(MN * 512 * 2);
  bf16* wq_b  = (bf16*)alloc((size_t)1536 * 512 * 2);
  bf16* wo_b  = (bf16*)alloc((size_t)512 * 512 * 2);
  bf16* qb    = (bf16*)alloc(BHND * 2);
  bf16* kb    = (bf16*)alloc(BHND * 2);
  bf16* vb    = (bf16*)alloc(BHND * 2);
  bf16* qlb   = (bf16*)alloc((size_t)32 * 256 * 64 * 2);
  bf16* klb   = (bf16*)alloc((size_t)32 * 256 * 64 * 2);
  float* attn2 = (float*)alloc((size_t)32 * 65536 * 4);
  bf16* abf   = (bf16*)alloc((size_t)32 * 65536 * 2);
  bf16* zA    = (bf16*)alloc((size_t)32 * 65536 * 2);
  bf16* zB    = (bf16*)alloc((size_t)32 * 65536 * 2);
  bf16* t0    = (bf16*)alloc((size_t)32 * 65536 * 2);
  bf16* t1    = (bf16*)alloc((size_t)32 * 65536 * 2);
  bf16* t2    = (bf16*)alloc((size_t)32 * 65536 * 2);
  float* bhmax = (float*)alloc(32 * 4);
  float* invrow = (float*)alloc(4);
  float* pm   = (float*)alloc((size_t)8 * 32 * 256 * 4);
  float* pl   = (float*)alloc((size_t)8 * 32 * 256 * 4);
  float* mfin = (float*)alloc((size_t)32 * 256 * 4);
  float* lfin = (float*)alloc((size_t)32 * 256 * 4);
  float* pO   = (float*)alloc((size_t)8 * 524288 * 4);
  bf16* av    = (bf16*)alloc((size_t)524288 * 2);
  bf16* Pb    = (bf16*)alloc((size_t)524288 * 2);
  bf16* Ptb   = (bf16*)alloc((size_t)524288 * 2);
  bf16* ctx   = (bf16*)alloc(MN * 512 * 2);

  // allow >48KB dynamic LDS (WGP has 320KB)
  hipFuncSetAttribute((const void*)gemm_xw_kernel,     hipFuncAttributeMaxDynamicSharedMemorySize, 73728);
  hipFuncSetAttribute((const void*)sim2_gemm_kernel,   hipFuncAttributeMaxDynamicSharedMemorySize, 55296);
  hipFuncSetAttribute((const void*)bgemm_kernel,       hipFuncAttributeMaxDynamicSharedMemorySize, 27648);
  hipFuncSetAttribute((const void*)flash3_stats_kernel,hipFuncAttributeMaxDynamicSharedMemorySize, 55296);
  hipFuncSetAttribute((const void*)flash3_av_kernel,   hipFuncAttributeMaxDynamicSharedMemorySize, 142336);
  hipFuncSetAttribute((const void*)flash1_kernel,      hipFuncAttributeMaxDynamicSharedMemorySize, 88064);

  ln_kernel<<<4096, 256, 0, stream>>>(x, gamma, beta, xn, 32768);
  cast_f32_to_bf16<<<3072, 256, 0, stream>>>(w_qkv, wq_b, 1536 * 512);
  cast_f32_to_bf16<<<1024, 256, 0, stream>>>(w_out, wo_b, 512 * 512);
  gemm_xw_kernel<<<dim3(12, 256), 256, 73728, stream>>>(xn, wq_b, qb, kb, vb,
                                                        nullptr, nullptr, nullptr, 0);
  landmark_kernel<<<dim3(32, 64), 256, 0, stream>>>(qb, kb, qlb, klb);
  sim2_gemm_kernel<<<dim3(2, 32), 256, 55296, stream>>>(qlb, klb, attn2);
  softmax256_kernel<<<32, 256, 0, stream>>>(attn2, abf, bhmax);
  rowscale_kernel<<<1, 32, 0, stream>>>(bhmax, invrow);
  zinit_kernel<<<8192, 256, 0, stream>>>(attn2, invrow, zA);
  flash3_stats_kernel<<<dim3(8, 32), 256, 55296, stream>>>(qlb, kb, pm, pl);
  stat_combine_kernel<<<32, 256, 0, stream>>>(pm, pl, mfin, lfin);
  flash3_av_kernel<<<dim3(8, 32), 256, 142336, stream>>>(qlb, kb, vb, mfin, lfin, pO);
  sum_partials_kernel<<<2048, 256, 0, stream>>>(pO, av);

  // Newton-Schulz pinv: z' = 3.25 z - 0.25 z@(15X - X@(7X - X@X)),  X = a@z
  bf16 *zc = zA, *zn = zB;
  for (int it = 0; it < 6; ++it) {
    bgemm_kernel<<<dim3(4, 2, 32), 256, 27648, stream>>>(abf, zc, t0, 256,  1.f,   0.f);
    bgemm_kernel<<<dim3(4, 2, 32), 256, 27648, stream>>>(t0,  t0, t1, 256, -1.f,   7.f);
    bgemm_kernel<<<dim3(4, 2, 32), 256, 27648, stream>>>(t0,  t1, t2, 256, -1.f,  15.f);
    bgemm_kernel<<<dim3(4, 2, 32), 256, 27648, stream>>>(zc,  t2, zn, 256, -0.25f, 3.25f);
    bf16* tmp = zc; zc = zn; zn = tmp;
  }
  bgemm_kernel<<<dim3(1, 2, 32), 256, 27648, stream>>>(zc, av, Pb, 64, 1.f, 0.f);
  transpose_p_kernel<<<2048, 256, 0, stream>>>(Pb, Ptb);
  flash1_kernel<<<dim3(64, 32), 256, 88064, stream>>>(qb, klb, Ptb, vb, w_res, ctx);
  gemm_xw_kernel<<<dim3(4, 256), 256, 73728, stream>>>(ctx, wo_b, nullptr, nullptr, nullptr,
                                                       x, b_out, out, 1);
}
